// ConvAttnPoolPlusConceptEmbeds_79044578116372
// MI455X (gfx1250) — matrix operand order repaired and checked by reading, not compiled, not run on hardware
//
#include <hip/hip_runtime.h>
#include <hip/hip_bf16.h>

// ---------------- problem constants ----------------
constexpr int B_    = 8;
constexpr int L_    = 2500;
constexpr int E_    = 100;
constexpr int F_    = 50;
constexpr int K_    = 10;
constexpr int Y_    = 8921;
constexpr int LOUT  = 2501;                 // conv output length (pad=5)
constexpr int ZROWS = 2544;                 // z_in rows: 5 left pad + 2500 + halo slack
constexpr int LP    = 2512;                 // padded L (mult of 16) for z_bf
constexpr int LT    = 2528;                 // padded L (mult of 32) for zT and LDS rows
constexpr int FP    = 64;                   // padded F (two 32-wide WMMA k-steps)
constexpr int YTILES = (Y_ + 15) / 16;      // 558

// workspace layout (all 256B aligned)
constexpr size_t Z_IN_BYTES = (size_t)B_ * ZROWS * E_ * sizeof(float);        // 8,140,800
constexpr size_t ZBF_BYTES  = (size_t)B_ * LP * FP * sizeof(unsigned short);  // 2,572,288
constexpr size_t ZT_BYTES   = (size_t)B_ * FP * LT * sizeof(unsigned short);  // 2,588,672

// ---------------- WMMA types ----------------
typedef __attribute__((ext_vector_type(16))) __bf16 v16bf;
typedef __attribute__((ext_vector_type(8)))  float  v8f;

union FragBF {
    v16bf v;
    unsigned short u[16];
    uint4 q[2];
};

__device__ __forceinline__ unsigned short f2bf(float x) {
    union { float f; unsigned u; } p; p.f = x;
    unsigned u = p.u;
    u += 0x7FFFu + ((u >> 16) & 1u);          // round-to-nearest-even
    return (unsigned short)(u >> 16);
}

// ---------------- kernel 1: embedding gather + concept select ----------------
__global__ __launch_bounds__(256)
void embed_kernel(const int* __restrict__ x, const int* __restrict__ concepts,
                  const float* __restrict__ embW, const float* __restrict__ conW,
                  float* __restrict__ z_in) {
    long idx = (long)blockIdx.x * blockDim.x + threadIdx.x;     // over B*ZROWS*E
    const long total = (long)B_ * ZROWS * E_;
    if (idx >= total) return;
    int e = (int)(idx % E_);
    int r = (int)((idx / E_) % ZROWS);
    int b = (int)(idx / ((long)E_ * ZROWS));
    float v = 0.f;
    if (r >= 5 && r < 5 + L_) {
        int l = r - 5;
        int c = concepts[(long)b * L_ + l];
        v = (c != 0) ? conW[(long)c * E_ + e]
                     : embW[(long)x[(long)b * L_ + l] * E_ + e];
    }
    z_in[idx] = v;
}

// ---------------- kernel 2: conv1d + bias + tanh -> bf16 (row-major and transposed)
__global__ __launch_bounds__(256)
void conv_kernel(const float* __restrict__ z_in, const float* __restrict__ conv_w,
                 const float* __restrict__ conv_b,
                 unsigned short* __restrict__ z_bf, unsigned short* __restrict__ zT) {
    __shared__ float win[41 * E_];            // input window rows t0 .. t0+40
    const int b  = blockIdx.y;
    const int t0 = blockIdx.x * 32;
    const int tid = threadIdx.x;

    for (int i = tid; i < 41 * E_; i += 256) {
        int r = i / E_, e = i % E_;
        win[i] = z_in[((long)b * ZROWS + (t0 + r)) * E_ + e];
    }
    __syncthreads();

    const int tl = tid & 31;                  // output position within chunk
    const int fg = tid >> 5;                  // 0..7 feature group
    const int t  = t0 + tl;
    if (t < LOUT) {
        for (int f = fg; f < F_; f += 8) {
            float acc = conv_b[f];
            const float* w = conv_w + (long)f * E_ * K_;   // [E][K]
            #pragma unroll
            for (int k = 0; k < K_; ++k) {
                const float* zr = &win[(tl + k) * E_];     // contiguous in e
                const float* wr = w + k;                   // stride K in e (cached)
                for (int e = 0; e < E_; ++e)
                    acc += zr[e] * wr[e * K_];
            }
            unsigned short hv = f2bf(tanhf(acc));
            z_bf[((long)b * LP + t) * FP + f] = hv;
            zT  [((long)b * FP + f) * LT + t] = hv;
        }
    }
    // zero-fill padded region (f>=50 or t>=2501); disjoint from compute writes
    for (int i = tid; i < 32 * FP; i += 256) {
        int tl2 = i / FP, f = i % FP;
        int tt = t0 + tl2;
        if (f >= F_ || tt >= LOUT) {
            if (tt < LP) z_bf[((long)b * LP + tt) * FP + f] = 0;
            if (tt < LT) zT  [((long)b * FP + f) * LT + tt] = 0;
        }
    }
}

// ---------------- kernel 3: fused per-label attention (WMMA) ----------------
// block = 128 threads (4 waves), one (b, 16-label tile) per block.
// dynamic LDS: 16 rows x LT halves: fp16 scores, overwritten with bf16 alpha (80,896 B)
__global__ __launch_bounds__(128)
void attnpool_kernel(const float* __restrict__ U_w, const float* __restrict__ final_w,
                     const float* __restrict__ final_b, const float* __restrict__ target,
                     const unsigned short* __restrict__ z_bf,
                     const unsigned short* __restrict__ zT,
                     float* __restrict__ d_yhat, float* __restrict__ d_alpha,
                     float* __restrict__ loss_partial) {
    extern __shared__ _Float16 sc[];          // [16][LT]
    __shared__ float red[128];
    __shared__ float row_max[16], row_sum[16];
    __shared__ float m_sh[16 * FP];
    __shared__ float loss_sh[16];

    const int b   = blockIdx.y;
    const int yt  = blockIdx.x;
    const int tid = threadIdx.x;
    const int lane = tid & 31;
    const int wv   = tid >> 5;                // wave id 0..3
    const int row  = lane & 15;               // A-matrix M row / C,B-matrix N
    const int hi   = lane >> 4;               // lane half select

    // --- build U A-fragments (16 labels x 32 f, two k-steps covering F=50->64)
    FragBF a0, a1;
    const int y_mine = yt * 16 + row;
    #pragma unroll
    for (int e = 0; e < 16; ++e) {
        int kk = (hi << 4) + e;                           // contiguous K = hi*16+e (matches B)
        int f0 = kk, f1 = 32 + kk;
        a0.u[e] = (y_mine < Y_ && f0 < F_) ? f2bf(U_w[(long)y_mine * F_ + f0]) : (unsigned short)0;
        a1.u[e] = (y_mine < Y_ && f1 < F_) ? f2bf(U_w[(long)y_mine * F_ + f1]) : (unsigned short)0;
    }

    // --- phase 1: scores[y, l] = sum_f U[y,f] * z[l,f]  -> LDS (fp16)
    for (int lt = wv; lt < LP / 16; lt += 4) {
        const int l = lt * 16 + row;          // B-matrix N = lane&15
        v8f c = {};
        FragBF fb;
        const uint4* p0 = (const uint4*)(z_bf + ((long)b * LP + l) * FP + hi * 16);
        fb.q[0] = p0[0]; fb.q[1] = p0[1];     // K = hi*16+e -> f 0..31
        c = __builtin_amdgcn_wmma_f32_16x16x32_bf16(false, a0.v, false, fb.v, (short)0, c, false, false);
        const uint4* p1 = (const uint4*)(z_bf + ((long)b * LP + l) * FP + 32 + hi * 16);
        fb.q[0] = p1[0]; fb.q[1] = p1[1];     // f 32..63
        c = __builtin_amdgcn_wmma_f32_16x16x32_bf16(false, a1.v, false, fb.v, (short)0, c, false, false);
        #pragma unroll
        for (int r = 0; r < 8; ++r)           // C layout: M = r + 8*hi, N = lane&15
            sc[(r + 8 * hi) * LT + lt * 16 + row] = (_Float16)c[r];
    }
    __syncthreads();

    // --- phase 2: row max / sum over l < 2501
    {
        const int r = tid & 15, ch = tid >> 4;            // 8 chunks of 316
        const int s = ch * 316;
        const int ep = (s + 316 < LOUT) ? s + 316 : LOUT;
        float mx = -3.4e38f;
        for (int i = s; i < ep; ++i) mx = fmaxf(mx, (float)sc[r * LT + i]);
        red[tid] = mx;
        __syncthreads();
        if (tid < 16) {
            float m2 = -3.4e38f;
            for (int c2 = 0; c2 < 8; ++c2) m2 = fmaxf(m2, red[c2 * 16 + tid]);
            row_max[tid] = m2;
        }
        __syncthreads();
        const float rm = row_max[r];
        float sm = 0.f;
        for (int i = s; i < ep; ++i) sm += expf((float)sc[r * LT + i] - rm);
        red[tid] = sm;
        __syncthreads();
        if (tid < 16) {
            float s2 = 0.f;
            for (int c2 = 0; c2 < 8; ++c2) s2 += red[c2 * 16 + tid];
            row_sum[tid] = s2;
        }
        __syncthreads();
    }

    // --- phase 3: write alpha to d_out (coalesced) and back to LDS as bf16 bits
    //              (tail l in [2501,LT) and invalid label rows zeroed)
    {
        unsigned short* au = (unsigned short*)sc;
        for (int yl = 0; yl < 16; ++yl) {
            const int  yy    = yt * 16 + yl;
            const bool valid = (yy < Y_);
            const float rm  = row_max[yl];
            const float inv = 1.f / row_sum[yl];
            const long base = ((long)b * Y_ + yy) * (long)LOUT;
            for (int l = tid; l < LT; l += 128) {
                float a = 0.f;
                if (valid && l < LOUT) {
                    a = expf((float)sc[yl * LT + l] - rm) * inv;
                    d_alpha[base + l] = a;
                }
                au[yl * LT + l] = f2bf(a);
            }
        }
    }
    __syncthreads();

    // --- phase 4: m[y, f] = sum_l alpha[y,l] * z[l,f]; wave wv owns f-tile wv
    //              A-fragment = two ds_load_b128 per lane (bf16 alpha in LDS)
    {
        v8f mc = {};
        const unsigned short* arow = (const unsigned short*)sc + row * LT;
        const int f = wv * 16 + row;          // B-matrix N
        const unsigned short* zrow = zT + ((long)b * FP + f) * LT;
        for (int l0 = 0; l0 < LT; l0 += 32) {
            FragBF fa, fb;
            // A layout: lane holds K = {hi*16..hi*16+15} (contiguous, matches B)
            fa.q[0] = *(const uint4*)(arow + l0 + hi * 16);
            fa.q[1] = *(const uint4*)(arow + l0 + hi * 16 + 8);
            const uint4* p = (const uint4*)(zrow + l0 + hi * 16);
            fb.q[0] = p[0]; fb.q[1] = p[1];   // K = hi*16+e -> l_local 0..31
            mc = __builtin_amdgcn_wmma_f32_16x16x32_bf16(false, fa.v, false, fb.v, (short)0, mc, false, false);
        }
        #pragma unroll
        for (int r = 0; r < 8; ++r)
            m_sh[(r + 8 * hi) * FP + wv * 16 + row] = mc[r];
    }
    __syncthreads();

    // --- phase 5: logits, sigmoid, BCE partials
    if (tid < 16) {
        const int yy = yt * 16 + tid;
        float lossv = 0.f;
        if (yy < Y_) {
            float z = final_b[yy];
            for (int f = 0; f < F_; ++f)
                z += final_w[(long)yy * F_ + f] * m_sh[tid * FP + f];
            d_yhat[(long)b * Y_ + yy] = 1.f / (1.f + expf(-z));
            const float t  = target[(long)b * Y_ + yy];
            const float sp = log1pf(expf(-fabsf(z)));
            const float sp_pos = sp + fmaxf(z, 0.f);      // softplus(z)
            const float sp_neg = sp + fmaxf(-z, 0.f);     // softplus(-z)
            lossv = t * sp_neg + (1.f - t) * sp_pos;
        }
        loss_sh[tid] = lossv;
    }
    __syncthreads();
    if (tid == 0) {
        float s = 0.f;
        for (int i = 0; i < 16; ++i) s += loss_sh[i];
        loss_partial[(long)b * YTILES + yt] = s;          // overwrite: replay-safe
    }
}

// ---------------- kernel 4: loss reduction ----------------
__global__ __launch_bounds__(256)
void loss_reduce(const float* __restrict__ partials, float* __restrict__ d_loss) {
    __shared__ float sh[256];
    float s = 0.f;
    for (int i = threadIdx.x; i < B_ * YTILES; i += 256) s += partials[i];
    sh[threadIdx.x] = s;
    __syncthreads();
    for (int st = 128; st > 0; st >>= 1) {
        if (threadIdx.x < st) sh[threadIdx.x] += sh[threadIdx.x + st];
        __syncthreads();
    }
    if (threadIdx.x == 0) d_loss[0] = sh[0] / (float)((long)B_ * Y_);
}

// ---------------- launch ----------------
extern "C" void kernel_launch(void* const* d_in, const int* in_sizes, int n_in,
                              void* d_out, int out_size, void* d_ws, size_t ws_size,
                              hipStream_t stream) {
    (void)in_sizes; (void)n_in; (void)out_size; (void)ws_size;
    const int*   x        = (const int*)  d_in[0];
    const int*   concepts = (const int*)  d_in[1];
    const float* target   = (const float*)d_in[2];
    const float* embW     = (const float*)d_in[3];
    const float* conW     = (const float*)d_in[4];
    const float* conv_w   = (const float*)d_in[5];
    const float* conv_b   = (const float*)d_in[6];
    const float* U_w      = (const float*)d_in[7];
    const float* final_w  = (const float*)d_in[8];
    const float* final_b  = (const float*)d_in[9];

    float* out    = (float*)d_out;
    float* d_yhat = out;
    float* d_loss = out + (long)B_ * Y_;
    float* d_alpha = out + (long)B_ * Y_ + 1;

    char* ws = (char*)d_ws;
    float*          z_in = (float*)ws;
    unsigned short* z_bf = (unsigned short*)(ws + Z_IN_BYTES);
    unsigned short* zT   = (unsigned short*)(ws + Z_IN_BYTES + ZBF_BYTES);
    float* partials      = (float*)(ws + Z_IN_BYTES + ZBF_BYTES + ZT_BYTES);

    {   // embedding
        long total = (long)B_ * ZROWS * E_;
        int blocks = (int)((total + 255) / 256);
        embed_kernel<<<blocks, 256, 0, stream>>>(x, concepts, embW, conW, z_in);
    }
    {   // conv + tanh + bf16 pack (row-major + transposed)
        dim3 grid(LT / 32, B_);               // 79 x 8, covers zero-fill to 2527
        conv_kernel<<<grid, 256, 0, stream>>>(z_in, conv_w, conv_b, z_bf, zT);
    }
    {   // fused attention: 558 label tiles x 8 batches, 4 waves each
        dim3 grid(YTILES, B_);
        size_t shmem = (size_t)16 * LT * sizeof(_Float16);   // 80,896 B
        attnpool_kernel<<<grid, 128, shmem, stream>>>(
            U_w, final_w, final_b, target, z_bf, zT, d_yhat, d_alpha, partials);
    }
    loss_reduce<<<1, 256, 0, stream>>>(partials, d_loss);
}
